// MvPDecoderLayer_38895223833163
// MI455X (gfx1250) — matrix operand order repaired and checked by reading, not compiled, not run on hardware
//
#include <hip/hip_runtime.h>
#include <hip/hip_bf16.h>
#include <math.h>

// ---------------------------------------------------------------------------
// MvP decoder layer for gfx1250 (MI455X). All GEMMs run through
// v_wmma_f32_16x16x32_f16 (wave32 WMMA). Memory-bound layer: f16 staging in
// LDS, f16 'val' tensor, flash attention (no NxN score matrix in HBM).
// Round 3: B tiles stream global->LDS via GLOBAL_LOAD_ASYNC_TO_LDS_B128
// (ASYNCcnt path, no VGPR round-trip); A stager uses b128 global loads +
// packed-f16 b128 LDS stores.
// ---------------------------------------------------------------------------

typedef __attribute__((ext_vector_type(16))) _Float16 v16h;
typedef __attribute__((ext_vector_type(8)))  _Float16 h8;
typedef __attribute__((ext_vector_type(8)))  float    v8f;

static constexpr int B_  = 4;
static constexpr int V_  = 5;
static constexpr int N_  = 1024;
static constexpr int D_  = 256;
static constexpr int NH_ = 8;
static constexpr int DH_ = 32;
static constexpr int DFF_ = 1024;
static constexpr int S_  = 10200;
static constexpr int BV_ = B_ * V_;

#define DEVFN static __device__ __forceinline__

DEVFN v8f wmma_f16(v16h a, v16h b, v8f c) {
  // D = A(16x32 f16) * B(32x16 f16) + C(16x16 f32)
  return __builtin_amdgcn_wmma_f32_16x16x32_f16(
      /*neg_a=*/false, a, /*neg_b=*/false, b,
      /*c_mod=*/(short)0, c, /*reuse_a=*/false, /*reuse_b=*/false);
}

DEVFN v16h mk16(h8 lo, h8 hi) {
  v16h r;
#pragma unroll
  for (int i = 0; i < 8; ++i) { r[i] = lo[i]; r[i + 8] = hi[i]; }
  return r;
}

// ---------------------------------------------------------------------------
// Generic WMMA GEMM: C[M,Nd] = A[M,K](f32, converted to f16 on the fly)
//                              @ WT[Nd,K](f16, pre-transposed) + bias,
// fused relu / row-mask / f16-output epilogue.
// Block: 128 threads (4 waves). Block tile 64x64, K-step 32.
// B tile copied with GLOBAL_LOAD_ASYNC_TO_LDS_B128 (ASYNCcnt).
// ---------------------------------------------------------------------------
__global__ __launch_bounds__(128) void gemm_wmma_kernel(
    const float* __restrict__ A, const _Float16* __restrict__ WT,
    const float* __restrict__ bias, float* __restrict__ Cf,
    _Float16* __restrict__ Ch, const unsigned char* __restrict__ rowmask,
    int M, int K, int Nd, int relu)
{
  __shared__ __align__(16) _Float16 As[64][40];   // [m][k], pad->16B aligned rows
  __shared__ __align__(16) _Float16 Bs[64][40];   // [n][k]

  const int tid  = threadIdx.x;
  const int lane = tid & 31;
  const int w    = tid >> 5;                      // wave id 0..3 -> 16-row strip
  const int m0   = blockIdx.x * 64;
  const int n0   = blockIdx.y * 64;

  v8f acc[4] = {};

  // A loader: 8 consecutive k per thread; rows ar0, ar0+32.
  const int af8 = tid & 3;                        // k chunk: 8 floats at af8*8
  const int ar0 = tid >> 2;                       // row base 0..31
  // B loader: 16 contiguous halves of WT per thread (one base, offset:16).
  const int bnn = tid & 63;                       // n row
  const int bkh = (tid >> 6) * 16;                // k half-group (0 or 16)
  // Hardware consumes addr[31:0] of a flat LDS pointer as the LDS address
  // (ISA 10.2 aperture translation), so truncate the pointer for VDST.
  const unsigned blds = (unsigned)(size_t)&Bs[bnn][bkh];

  // Clamp rows once (branch-free); out-of-range rows are write-guarded later.
  int agm[2];
#pragma unroll
  for (int j = 0; j < 2; ++j) {
    int gm = m0 + ar0 + j * 32;
    agm[j] = gm < M ? gm : M - 1;
  }

  for (int k0 = 0; k0 < K; k0 += 32) {
    // B tile: async copy straight into LDS (no VGPR staging). INST_OFFSET is
    // added to BOTH the LDS and global addresses, so one base serves 32B.
    {
      const _Float16* wrow = WT + (size_t)(n0 + bnn) * K + k0 + bkh;
      asm volatile(
          "global_load_async_to_lds_b128 %0, %1, off\n\t"
          "global_load_async_to_lds_b128 %0, %1, off offset:16"
          :
          : "v"(blds), "v"(wrow)
          : "memory");
    }
    if (k0 + 32 < K)                              // speculative global_prefetch_b8
      __builtin_prefetch(&A[(size_t)agm[0] * K + k0 + 32 + af8 * 8], 0, 1);
    // A tile: two b128 global loads -> packed f16 -> one ds_store_b128 per row.
#pragma unroll
    for (int j = 0; j < 2; ++j) {
      const float4 f0 = *(const float4*)&A[(size_t)agm[j] * K + k0 + af8 * 8];
      const float4 f1 = *(const float4*)&A[(size_t)agm[j] * K + k0 + af8 * 8 + 4];
      h8 hv = {(_Float16)f0.x, (_Float16)f0.y, (_Float16)f0.z, (_Float16)f0.w,
               (_Float16)f1.x, (_Float16)f1.y, (_Float16)f1.z, (_Float16)f1.w};
      *(h8*)&As[ar0 + j * 32][af8 * 8] = hv;      // ds_store_b128
    }
    asm volatile("s_wait_asynccnt 0x0" ::: "memory");
    __syncthreads();

    const int fr = lane & 15;
    const int kb = (lane >> 4) * 8;    // A frag: lanes<16 -> K{0..7,16..23}; >=16 -> K{8..15,24..31}
    const int ks = (lane >> 4) * 8;    // B frag: lanes<16 -> K{0..7,16..23}; >=16 -> K{8..15,24..31}
    v16h a = mk16(*(const h8*)&As[w * 16 + fr][kb],
                  *(const h8*)&As[w * 16 + fr][kb + 16]);
#pragma unroll
    for (int t = 0; t < 4; ++t) {
      v16h b = mk16(*(const h8*)&Bs[t * 16 + fr][ks],
                    *(const h8*)&Bs[t * 16 + fr][ks + 16]);
      acc[t] = wmma_f16(a, b, acc[t]);
    }
    __syncthreads();
  }

  // C/D layout: lane<16 -> (m=r, n=lane); lane>=16 -> (m=r+8, n=lane-16)
  const int rh = 8 * (lane >> 4);
  const int nn = lane & 15;
#pragma unroll
  for (int t = 0; t < 4; ++t)
#pragma unroll
    for (int r = 0; r < 8; ++r) {
      int m = m0 + w * 16 + r + rh;
      int n = n0 + t * 16 + nn;
      if (m < M) {
        float vv = acc[t][r] + (bias ? bias[n] : 0.f);
        if (relu) vv = fmaxf(vv, 0.f);
        if (rowmask && rowmask[m]) vv = 0.f;   // fused src_padding_mask
        if (Cf) Cf[(size_t)m * Nd + n] = vv;
        else    Ch[(size_t)m * Nd + n] = (_Float16)vv;
      }
    }
}

// ---------------------------------------------------------------------------
// Flash attention: one wave per (b, h, 16-query tile). DH=32 => Q tile is a
// single WMMA A operand. Online softmax, 32 keys per step (2 score WMMAs +
// 2 AV WMMAs). V is pre-transposed to [b][h][dh][key] so every fragment load
// is an aligned 16-byte (h8) load.
// ---------------------------------------------------------------------------
__global__ __launch_bounds__(32) void attn_flash_kernel(
    const _Float16* __restrict__ q16, const _Float16* __restrict__ k16,
    const _Float16* __restrict__ vt, float* __restrict__ ao)
{
  constexpr float SCALE = 0.17677669529663687f;   // 1/sqrt(32)
  const int wg = blockIdx.x;
  const int qt = wg & 63;              // N/16 = 64 query tiles
  const int h  = (wg >> 6) & 7;
  const int b  = wg >> 9;
  const int lane = threadIdx.x;
  const int qm = lane & 15, hi = lane >> 4;

  __shared__ __align__(16) float    Ss[16][36];
  __shared__ __align__(16) _Float16 Ps[16][40];
  __shared__ float mrow[16], lrow[16], srow[16];

  const size_t qbase = ((size_t)(b * N_) + qt * 16 + qm) * D_ + h * DH_;
  v16h aq = mk16(*(const h8*)(q16 + qbase + hi * 8),
                 *(const h8*)(q16 + qbase + hi * 8 + 16));

  if (lane < 16) { mrow[lane] = -3.0e38f; lrow[lane] = 0.f; }
  v8f c0 = {}, c1 = {};
  __syncthreads();

  for (int kt = 0; kt < N_ / 32; ++kt) {
    const int key0 = kt * 32;
    // ---- scores: S = Q * K^T for 32 keys (two 16-key WMMAs) ----
#pragma unroll
    for (int hf = 0; hf < 2; ++hf) {
      const size_t kbase =
          ((size_t)(b * N_) + key0 + hf * 16 + qm) * D_ + h * DH_ + hi * 8;
      v16h bk = mk16(*(const h8*)(k16 + kbase), *(const h8*)(k16 + kbase + 16));
      v8f s = {};
      s = wmma_f16(aq, bk, s);
#pragma unroll
      for (int r = 0; r < 8; ++r)
        Ss[r + 8 * hi][hf * 16 + qm] = s[r] * SCALE;
    }
    __syncthreads();
    // ---- online softmax (lanes 0..15 each own one query row) ----
    if (lane < 16) {
      const int row = lane;
      float mo = mrow[row], tm = mo;
      for (int j = 0; j < 32; ++j) tm = fmaxf(tm, Ss[row][j]);
      float sc = __expf(mo - tm), sum = 0.f;
      for (int j = 0; j < 32; ++j) {
        float pj = __expf(Ss[row][j] - tm);
        Ps[row][j] = (_Float16)pj;
        sum += pj;
      }
      mrow[row] = tm;
      lrow[row] = lrow[row] * sc + sum;
      srow[row] = sc;
    }
    __syncthreads();
#pragma unroll
    for (int r = 0; r < 8; ++r) {
      float sc = srow[r + 8 * hi];
      c0[r] *= sc; c1[r] *= sc;
    }
    // ---- AV: P(16x32) * V(32x32) as two 32x16 B operands ----
    v16h ap = mk16(*(const h8*)&Ps[qm][hi * 8],
                   *(const h8*)&Ps[qm][hi * 8 + 16]);
#pragma unroll
    for (int t = 0; t < 2; ++t) {
      // B frag element e -> key0 + hi*8 + e (e<8), key0 + 16 + hi*8 + (e-8) (e>=8), n = dh = t*16+qm; vt is key-major
      const size_t vbase =
          (((size_t)(b * NH_) + h) * DH_ + t * 16 + qm) * N_ + key0 + hi * 8;
      v16h bvv = mk16(*(const h8*)(vt + vbase), *(const h8*)(vt + vbase + 16));
      if (t == 0) c0 = wmma_f16(ap, bvv, c0);
      else        c1 = wmma_f16(ap, bvv, c1);
    }
    __syncthreads();
  }

#pragma unroll
  for (int r = 0; r < 8; ++r) {
    const int m = r + 8 * hi;
    const float inv = 1.f / lrow[m];
    const size_t ob = ((size_t)(b * N_) + qt * 16 + m) * D_ + h * DH_;
    ao[ob + qm]      = c0[r] * inv;
    ao[ob + 16 + qm] = c1[r] * inv;
  }
}

// --------------------------- small VALU kernels ----------------------------
// f32 -> f16 with transpose: WT[n][k] = W[k][n] (writes coalesced).
__global__ void cvtT_kernel(const float* __restrict__ W,
                            _Float16* __restrict__ WT, int K, int Nd) {
  int i = blockIdx.x * blockDim.x + threadIdx.x;
  if (i >= K * Nd) return;
  int k = i % K, n = i / K;
  WT[i] = (_Float16)W[(size_t)k * Nd + n];
}

// v16[b,n,h,dh] -> vt[b,h,dh,n] (writes coalesced).
__global__ void vtrans_kernel(const _Float16* __restrict__ v16,
                              _Float16* __restrict__ vt) {
  int i = blockIdx.x * blockDim.x + threadIdx.x;
  if (i >= B_ * N_ * D_) return;
  int key = i % N_;
  int dh  = (i / N_) % DH_;
  int h   = (i / (N_ * DH_)) % NH_;
  int b   = i / (N_ * DH_ * NH_);
  vt[i] = v16[((size_t)(b * N_) + key) * D_ + h * DH_ + dh];
}

__global__ void add2_kernel(const float* __restrict__ a,
                            const float* __restrict__ b,
                            float* __restrict__ c, int n) {
  int i = blockIdx.x * blockDim.x + threadIdx.x;
  if (i < n) c[i] = a[i] + b[i];
}

// qpa[bv,n,d] = target2[b,n,d] + query_pos[b,n,d]  (broadcast over V)
__global__ void qpa_kernel(const float* __restrict__ t2,
                           const float* __restrict__ qp,
                           float* __restrict__ qpa) {
  int i = blockIdx.x * blockDim.x + threadIdx.x;
  if (i >= BV_ * N_ * D_) return;
  int d  = i % D_;
  int n  = (i / D_) % N_;
  int bv = i / (D_ * N_);
  int b  = bv / V_;
  size_t src = ((size_t)(b * N_) + n) * D_ + d;
  qpa[i] = t2[src] + qp[src];
}

// Camera projection: ref2d + bounding per (b,v,n)
__global__ void geom_kernel(const float* __restrict__ ref,
                            const float* __restrict__ camK,
                            const float* __restrict__ camR,
                            const float* __restrict__ camT,
                            const float* __restrict__ trans,
                            const float* __restrict__ wh,
                            float* __restrict__ ref2d,
                            float* __restrict__ bound) {
  int idx = blockIdx.x * blockDim.x + threadIdx.x;
  if (idx >= BV_ * N_) return;
  int n = idx % N_, bv = idx / N_, b = bv / V_;
  const float GS[3] = {8000.f, 8000.f, 2000.f};
  const float GC[3] = {0.f, 0.f, 1000.f};
  float ra[3];
#pragma unroll
  for (int j = 0; j < 3; ++j)
    ra[j] = (ref[((size_t)(b * N_) + n) * 3 + j] - 0.5f) * GS[j] + GC[j];
  const float* R = camR + (size_t)bv * 9;
  const float* T = camT + (size_t)bv * 3;
  float Xc[3];
#pragma unroll
  for (int i = 0; i < 3; ++i)
    Xc[i] = R[i * 3 + 0] * (ra[0] - T[0]) + R[i * 3 + 1] * (ra[1] - T[1]) +
            R[i * 3 + 2] * (ra[2] - T[2]);
  float invz = 1.f / Xc[2];
  const float* Kc = camK + (size_t)bv * 9;
  float px0 = Xc[0] * invz * Kc[0] + Kc[2];
  float px1 = Xc[1] * invz * Kc[4] + Kc[5];
  float w0 = wh[(size_t)bv * 2 + 0], w1 = wh[(size_t)bv * 2 + 1];
  float bnd = (px0 >= 0.f && px1 >= 0.f && px0 < w0 && px1 < w1) ? 1.f : 0.f;
  float mx = fmaxf(w0, w1);                 // wh.max() (uniform wh in this model)
  px0 = fminf(fmaxf(px0, -1.f), mx);
  px1 = fminf(fmaxf(px1, -1.f), mx);
  const float* tr = trans + (size_t)bv * 6;
  float tx = tr[0] * px0 + tr[1] * px1 + tr[2];
  float ty = tr[3] * px0 + tr[4] * px1 + tr[5];
  ref2d[(size_t)idx * 2 + 0] = tx / 960.f;  // IMG_SIZE
  ref2d[(size_t)idx * 2 + 1] = ty / 512.f;
  bound[idx] = bnd;
}

// softmax over L*P=16 per (bv,n,h)
__global__ void awsoftmax_kernel(const float* __restrict__ logits,
                                 float* __restrict__ aw) {
  int idx = blockIdx.x * blockDim.x + threadIdx.x;
  if (idx >= BV_ * N_ * NH_) return;
  int h = idx % NH_;
  size_t bn = idx / NH_;
  const float* src = logits + bn * (NH_ * 16) + h * 16;
  float mx = src[0];
  for (int j = 1; j < 16; ++j) mx = fmaxf(mx, src[j]);
  float e[16], sum = 0.f;
  for (int j = 0; j < 16; ++j) { e[j] = __expf(src[j] - mx); sum += e[j]; }
  float inv = 1.f / sum;
  float* dst = aw + bn * (NH_ * 16) + h * 16;
  for (int j = 0; j < 16; ++j) dst[j] = e[j] * inv;
}

// Deformable bilinear sampling: one wave per (bv,n,h); lane = channel (DH=32).
__global__ __launch_bounds__(256) void deform_kernel(
    const _Float16* __restrict__ val16, const float* __restrict__ off,
    const float* __restrict__ aw, const float* __restrict__ ref2d,
    float* __restrict__ dout)
{
  const int wid  = blockIdx.x * 8 + (threadIdx.x >> 5);
  const int lane = threadIdx.x & 31;
  if (wid >= BV_ * N_ * NH_) return;
  const int h  = wid % NH_;
  const int n  = (wid / NH_) % N_;
  const int bv = wid / (NH_ * N_);
  const size_t bn = (size_t)bv * N_ + n;
  const float rx = ref2d[bn * 2 + 0], ry = ref2d[bn * 2 + 1];
  const int LH[4] = {64, 32, 16, 8};
  const int LW[4] = {120, 60, 30, 15};
  const int LST[4] = {0, 7680, 9600, 10080};
  float acc = 0.f;
#pragma unroll
  for (int l = 0; l < 4; ++l) {
    const float sfx = (float)LW[l], sfy = (float)LH[l];
    const float rex = rx * sfx / (sfx - 1.f);
    const float rey = ry * sfy / (sfy - 1.f);
    const size_t lvlbase = (size_t)bv * S_ + LST[l];
#pragma unroll
    for (int p = 0; p < 4; ++p) {
      const size_t ob = bn * 256 + (size_t)(((h * 4 + l) * 4 + p) * 2);
      float lx = rex + off[ob + 0] / sfx;
      float ly = rey + off[ob + 1] / sfy;
      float pxx = lx * sfx - 0.5f, pyy = ly * sfy - 0.5f;
      float x0f = floorf(pxx), y0f = floorf(pyy);
      float fx = pxx - x0f, fy = pyy - y0f;
      int ix = (int)x0f, iy = (int)y0f;
      float a = aw[bn * 128 + h * 16 + l * 4 + p];
      auto samp = [&](int xx, int yy) -> float {
        if (xx < 0 || yy < 0 || xx >= LW[l] || yy >= LH[l]) return 0.f;
        return (float)val16[(lvlbase + (size_t)yy * LW[l] + xx) * D_ +
                            h * DH_ + lane];
      };
      float v00 = samp(ix, iy),     v10 = samp(ix + 1, iy);
      float v01 = samp(ix, iy + 1), v11 = samp(ix + 1, iy + 1);
      acc += a * (v00 * (1.f - fx) * (1.f - fy) + v10 * fx * (1.f - fy) +
                  v01 * (1.f - fx) * fy + v11 * fx * fy);
    }
  }
  dout[bn * D_ + h * DH_ + lane] = acc;
}

// tt[b,n,d] = mean over V of bounding * pa
__global__ void vmean_kernel(const float* __restrict__ pa,
                             const float* __restrict__ bound,
                             float* __restrict__ tt) {
  const int d = threadIdx.x;
  const int n = blockIdx.x % N_, b = blockIdx.x / N_;
  float s = 0.f;
#pragma unroll
  for (int v = 0; v < V_; ++v) {
    const int bv = b * V_ + v;
    s += bound[(size_t)bv * N_ + n] * pa[((size_t)bv * N_ + n) * D_ + d];
  }
  tt[((size_t)(b * N_) + n) * D_ + d] = s * (1.f / V_);
}

// out = LayerNorm(x + r) over last dim (D=256), one row per 256-thread block
__global__ __launch_bounds__(256) void add_ln_kernel(
    const float* __restrict__ x, const float* __restrict__ r,
    const float* __restrict__ g, const float* __restrict__ bb,
    float* __restrict__ out)
{
  __shared__ float red[256];
  const int t = threadIdx.x;
  const size_t row = blockIdx.x;
  float v = x[row * D_ + t] + r[row * D_ + t];
  red[t] = v; __syncthreads();
  for (int s = 128; s > 0; s >>= 1) { if (t < s) red[t] += red[t + s]; __syncthreads(); }
  float mean = red[0] * (1.f / D_); __syncthreads();
  float dlt = v - mean;
  red[t] = dlt * dlt; __syncthreads();
  for (int s = 128; s > 0; s >>= 1) { if (t < s) red[t] += red[t + s]; __syncthreads(); }
  float var = red[0] * (1.f / D_);
  out[row * D_ + t] = dlt * rsqrtf(var + 1e-5f) * g[t] + bb[t];
}

// ---------------------------------------------------------------------------
extern "C" void kernel_launch(void* const* d_in, const int* in_sizes, int n_in,
                              void* d_out, int out_size, void* d_ws, size_t ws_size,
                              hipStream_t stream)
{
  (void)in_sizes; (void)n_in; (void)out_size; (void)ws_size;

  const float* target    = (const float*)d_in[0];
  const float* query_pos = (const float*)d_in[1];
  const float* refpts    = (const float*)d_in[2];
  const float* src_flat  = (const float*)d_in[3];
  // d_in[4] src_spatial_shapes, d_in[5] level_start_index: compile-time consts
  const float* camK  = (const float*)d_in[6];
  const float* camR  = (const float*)d_in[7];
  const float* camT  = (const float*)d_in[8];
  const float* trans = (const float*)d_in[9];
  const float* wh    = (const float*)d_in[10];
  const unsigned char* padmask = (const unsigned char*)d_in[11]; // bool mask

  // params dict in insertion order
  const float* Wq = (const float*)d_in[12]; const float* bq = (const float*)d_in[13];
  const float* Wk = (const float*)d_in[14]; const float* bk = (const float*)d_in[15];
  const float* Wv = (const float*)d_in[16]; const float* bvb = (const float*)d_in[17];
  const float* Wo = (const float*)d_in[18]; const float* bo = (const float*)d_in[19];
  const float* ln1g = (const float*)d_in[20]; const float* ln1b = (const float*)d_in[21];
  const float* ln2g = (const float*)d_in[22]; const float* ln2b = (const float*)d_in[23];
  const float* ln3g = (const float*)d_in[24]; const float* ln3b = (const float*)d_in[25];
  const float* Woff = (const float*)d_in[26]; const float* boff = (const float*)d_in[27];
  const float* Wattn = (const float*)d_in[28]; const float* battn = (const float*)d_in[29];
  const float* Wval = (const float*)d_in[30]; const float* bval = (const float*)d_in[31];
  const float* Wpo  = (const float*)d_in[32]; const float* bpo  = (const float*)d_in[33];
  const float* W1 = (const float*)d_in[34]; const float* b1 = (const float*)d_in[35];
  const float* W2 = (const float*)d_in[36]; const float* b2 = (const float*)d_in[37];

  // ---- workspace bump allocator (~266 MB total) ----
  char* p = (char*)d_ws;
  auto alloc = [&](size_t bytes) -> void* {
    void* r = p; p += (bytes + 255) & ~(size_t)255; return r;
  };
  _Float16* wq16   = (_Float16*)alloc((size_t)D_ * D_ * 2);
  _Float16* wk16   = (_Float16*)alloc((size_t)D_ * D_ * 2);
  _Float16* wv16   = (_Float16*)alloc((size_t)D_ * D_ * 2);
  _Float16* wo16   = (_Float16*)alloc((size_t)D_ * D_ * 2);
  _Float16* woff16 = (_Float16*)alloc((size_t)D_ * 256 * 2);
  _Float16* wattn16= (_Float16*)alloc((size_t)D_ * 128 * 2);
  _Float16* wval16 = (_Float16*)alloc((size_t)D_ * D_ * 2);
  _Float16* wpo16  = (_Float16*)alloc((size_t)D_ * D_ * 2);
  _Float16* w1_16  = (_Float16*)alloc((size_t)D_ * DFF_ * 2);
  _Float16* w2_16  = (_Float16*)alloc((size_t)DFF_ * D_ * 2);

  float*    qkbuf  = (float*)alloc((size_t)B_ * N_ * D_ * 4);
  _Float16* q16    = (_Float16*)alloc((size_t)B_ * N_ * D_ * 2);
  _Float16* k16b   = (_Float16*)alloc((size_t)B_ * N_ * D_ * 2);
  _Float16* v16b   = (_Float16*)alloc((size_t)B_ * N_ * D_ * 2);
  _Float16* vtb    = (_Float16*)alloc((size_t)B_ * N_ * D_ * 2);
  float*    ao     = (float*)alloc((size_t)B_ * N_ * D_ * 4);
  float*    sa     = (float*)alloc((size_t)B_ * N_ * D_ * 4);
  float*    target2= (float*)alloc((size_t)B_ * N_ * D_ * 4);
  float*    ref2d  = (float*)alloc((size_t)BV_ * N_ * 2 * 4);
  float*    bound  = (float*)alloc((size_t)BV_ * N_ * 4);
  float*    qpaB   = (float*)alloc((size_t)BV_ * N_ * D_ * 4);
  float*    offB   = (float*)alloc((size_t)BV_ * N_ * 256 * 4);
  float*    logits = (float*)alloc((size_t)BV_ * N_ * 128 * 4);
  float*    awB    = (float*)alloc((size_t)BV_ * N_ * 128 * 4);
  _Float16* val16B = (_Float16*)alloc((size_t)BV_ * S_ * D_ * 2);   // 104 MB
  float*    doutB  = (float*)alloc((size_t)BV_ * N_ * D_ * 4);
  float*    paB    = (float*)alloc((size_t)BV_ * N_ * D_ * 4);
  float*    ttB    = (float*)alloc((size_t)B_ * N_ * D_ * 4);
  float*    target3= (float*)alloc((size_t)B_ * N_ * D_ * 4);
  float*    h1B    = (float*)alloc((size_t)B_ * N_ * DFF_ * 4);
  float*    h2B    = (float*)alloc((size_t)B_ * N_ * D_ * 4);

  // ---- convert + transpose weights to f16 once per call: WT[n][k] ----
  struct CvtJob { const float* s; _Float16* d; int K; int Nd; } jobs[10] = {
    {Wq, wq16, D_, D_}, {Wk, wk16, D_, D_}, {Wv, wv16, D_, D_},
    {Wo, wo16, D_, D_}, {Woff, woff16, D_, 256}, {Wattn, wattn16, D_, 128},
    {Wval, wval16, D_, D_}, {Wpo, wpo16, D_, D_},
    {W1, w1_16, D_, DFF_}, {W2, w2_16, DFF_, D_},
  };
  for (int i = 0; i < 10; ++i) {
    int n = jobs[i].K * jobs[i].Nd;
    cvtT_kernel<<<(n + 255) / 256, 256, 0, stream>>>(jobs[i].s, jobs[i].d,
                                                     jobs[i].K, jobs[i].Nd);
  }

  auto gemm = [&](const float* A, const _Float16* Wt, const float* bias,
                  float* Cf, _Float16* Ch, const unsigned char* rm,
                  int M, int K, int Nd, int relu) {
    dim3 g((unsigned)((M + 63) / 64), (unsigned)(Nd / 64));
    gemm_wmma_kernel<<<g, 128, 0, stream>>>(A, Wt, bias, Cf, Ch, rm, M, K, Nd, relu);
  };

  // ---- self attention ----
  add2_kernel<<<(B_ * N_ * D_ + 255) / 256, 256, 0, stream>>>(
      target, query_pos, qkbuf, B_ * N_ * D_);
  gemm(qkbuf,  wq16, bq,  nullptr, q16,  nullptr, B_ * N_, D_, D_, 0);
  gemm(qkbuf,  wk16, bk,  nullptr, k16b, nullptr, B_ * N_, D_, D_, 0);
  gemm(target, wv16, bvb, nullptr, v16b, nullptr, B_ * N_, D_, D_, 0);
  vtrans_kernel<<<(B_ * N_ * D_ + 255) / 256, 256, 0, stream>>>(v16b, vtb);
  attn_flash_kernel<<<B_ * NH_ * (N_ / 16), 32, 0, stream>>>(q16, k16b, vtb, ao);
  gemm(ao, wo16, bo, sa, nullptr, nullptr, B_ * N_, D_, D_, 0);
  add_ln_kernel<<<B_ * N_, 256, 0, stream>>>(target, sa, ln2g, ln2b, target2);

  // ---- camera geometry + deformable cross attention ----
  geom_kernel<<<(BV_ * N_ + 255) / 256, 256, 0, stream>>>(
      refpts, camK, camR, camT, trans, wh, ref2d, bound);
  qpa_kernel<<<(BV_ * N_ * D_ + 255) / 256, 256, 0, stream>>>(target2, query_pos, qpaB);
  gemm(qpaB, woff16,  boff,  offB,   nullptr, nullptr, BV_ * N_, D_, 256, 0);
  gemm(qpaB, wattn16, battn, logits, nullptr, nullptr, BV_ * N_, D_, 128, 0);
  awsoftmax_kernel<<<(BV_ * N_ * NH_ + 255) / 256, 256, 0, stream>>>(logits, awB);
  gemm(src_flat, wval16, bval, nullptr, val16B, padmask, BV_ * S_, D_, D_, 0); // dominant GEMM
  deform_kernel<<<(BV_ * N_ * NH_) / 8, 256, 0, stream>>>(val16B, offB, awB, ref2d, doutB);
  gemm(doutB, wpo16, bpo, paB, nullptr, nullptr, BV_ * N_, D_, D_, 0);
  vmean_kernel<<<B_ * N_, 256, 0, stream>>>(paB, bound, ttB);
  add_ln_kernel<<<B_ * N_, 256, 0, stream>>>(target2, ttB, ln1g, ln1b, target3);

  // ---- FFN ----
  gemm(target3, w1_16, b1, h1B, nullptr, nullptr, B_ * N_, D_, DFF_, 1);
  gemm(h1B, w2_16, b2, h2B, nullptr, nullptr, B_ * N_, DFF_, D_, 0);
  add_ln_kernel<<<B_ * N_, 256, 0, stream>>>(target3, h2B, ln3g, ln3b, (float*)d_out);
}
